// _NonLocalBlockND_50113678409821
// MI455X (gfx1250) — matrix operand order repaired and checked by reading, not compiled, not run on hardware
//
#include <hip/hip_runtime.h>
#include <hip/hip_bf16.h>
#include <cstdint>

#define BATCH 4
#define C_IN  256
#define CI_   128
#define HW_   4096
#define EPS_  1e-5f

#ifndef NO_ASYNC_LDS
#if __has_builtin(__builtin_amdgcn_global_load_async_to_lds_b128)
#define USE_ASYNC_LDS 1
#else
#define USE_ASYNC_LDS 0
#endif
#else
#define USE_ASYNC_LDS 0
#endif

typedef __attribute__((ext_vector_type(16))) __bf16 v16bf;
typedef __attribute__((ext_vector_type(8)))  float  v8f;
typedef int v4i_ __attribute__((vector_size(16)));

union FragU { v16bf v; uint32_t u[8]; };
struct AccT { union { v8f v; float f[8]; }; };

__device__ __forceinline__ uint16_t f2bf(float f) {
  uint32_t x = __float_as_uint(f);
  uint32_t r = x + 0x7FFFu + ((x >> 16) & 1u);   // round-to-nearest-even
  return (uint16_t)(r >> 16);
}
__device__ __forceinline__ uint32_t f2bf2(float lo, float hi) {
  return (uint32_t)f2bf(lo) | ((uint32_t)f2bf(hi) << 16);
}

// 16-byte global->LDS copy; async (ASYNCcnt) when the builtin exists.
__device__ __forceinline__ void copy16_g2l(const uint16_t* gsrc, uint16_t* ldst) {
#if USE_ASYNC_LDS
  __builtin_amdgcn_global_load_async_to_lds_b128(
      (__attribute__((address_space(1))) v4i_*)(uintptr_t)gsrc,
      (__attribute__((address_space(3))) v4i_*)(uintptr_t)ldst, 0, 0);
#else
  *(uint4*)ldst = *(const uint4*)gsrc;
#endif
}
__device__ __forceinline__ void copy_join() {
#if USE_ASYNC_LDS
#if __has_builtin(__builtin_amdgcn_s_wait_asynccnt)
  __builtin_amdgcn_s_wait_asynccnt(0);
#else
  asm volatile("s_wait_asynccnt 0x0" ::: "memory");
#endif
#endif
}

// A fragment: 16x32 (MxK) bf16, source tile row-major [m][k], k contiguous.
__device__ __forceinline__ v16bf load_frag_a(const uint16_t* t, int ld) {
  int lane = threadIdx.x & 31;
  int m = lane & 15, half = lane >> 4;
  FragU f;
#pragma unroll
  for (int v = 0; v < 8; ++v) {
    int k = ((v >> 2) << 4) + (half << 3) + ((v & 3) << 1);
    f.u[v] = *(const uint32_t*)(t + m * ld + k);
  }
  return f.v;
}

// B fragment: 32x16 (KxN) bf16, source tile stored [n][k], k contiguous.
__device__ __forceinline__ v16bf load_frag_b(const uint16_t* t, int ld) {
  int lane = threadIdx.x & 31;
  int n = lane & 15, half = lane >> 4;
  FragU f;
#pragma unroll
  for (int v = 0; v < 8; ++v) {
    int k = ((v >> 2) << 4) + (half << 3) + ((v & 3) << 1);
    f.u[v] = *(const uint32_t*)(t + n * ld + k);
  }
  return f.v;
}

__device__ __forceinline__ v8f wmma_bf16(v16bf a, v16bf b, v8f c) {
  return __builtin_amdgcn_wmma_f32_16x16x32_bf16(false, a, false, b,
                                                 (short)0, c, false, false);
}

// ---------------------------------------------------------------------------
// Kernel 1: five 1x1-conv projections.
// p=0 (g): output [CI][N] (natural).  p=1..4 (theta/phi/h_theta/h_phi):
// output TRANSPOSED [N][CI] so attention staging is contiguous.
// ---------------------------------------------------------------------------
__global__ __launch_bounds__(256) void proj_kernel(
    const float* __restrict__ feats, const float* __restrict__ human,
    const float* __restrict__ g_w,  const float* __restrict__ g_b,
    const float* __restrict__ t_w,  const float* __restrict__ t_b,
    const float* __restrict__ p_w,  const float* __restrict__ p_b,
    const float* __restrict__ ht_w, const float* __restrict__ ht_b,
    const float* __restrict__ hp_w, const float* __restrict__ hp_b,
    uint16_t* __restrict__ ws) {
  const int nb = blockIdx.x, b = blockIdx.y, p = blockIdx.z;
  const float *w, *bias, *src;
  switch (p) {
    case 0: w = g_w;  bias = g_b;  src = feats; break;
    case 1: w = t_w;  bias = t_b;  src = feats; break;
    case 2: w = p_w;  bias = p_b;  src = feats; break;
    case 3: w = ht_w; bias = ht_b; src = human; break;
    default: w = hp_w; bias = hp_b; src = human; break;
  }
  uint16_t* dst = ws + ((size_t)p * BATCH + b) * CI_ * HW_;
  src += (size_t)b * C_IN * HW_;
  const int n0 = nb * 128;

  __shared__ __attribute__((aligned(16))) uint16_t Wt[128 * 32];
  __shared__ __attribute__((aligned(16))) uint16_t Xt[128 * 34];

  const int wave = threadIdx.x >> 5;
  const int m0 = (wave & 3) * 32, n0w = (wave >> 2) * 64;

  AccT acc[2][4];
#pragma unroll
  for (int i = 0; i < 2; ++i)
#pragma unroll
    for (int j = 0; j < 4; ++j)
#pragma unroll
      for (int e = 0; e < 8; ++e) acc[i][j].f[e] = 0.f;

  for (int kb = 0; kb < 8; ++kb) {
    // weights: [m][k], k contiguous both sides -> vectorized
    for (int i = threadIdx.x; i < 1024; i += 256) {
      int m = i >> 3, k4 = (i & 7) * 4;
      float4 wv = *(const float4*)(w + m * C_IN + kb * 32 + k4);
      uint2 pk; pk.x = f2bf2(wv.x, wv.y); pk.y = f2bf2(wv.z, wv.w);
      *(uint2*)(Wt + m * 32 + k4) = pk;
    }
    // activations: [c][n] -> LDS [n][k]; vector load, scattered bf16 stores
    for (int i = threadIdx.x; i < 1024; i += 256) {
      int k = i >> 5, n4 = (i & 31) * 4;
      float4 xv = *(const float4*)(src + (size_t)(kb * 32 + k) * HW_ + n0 + n4);
      Xt[(n4 + 0) * 34 + k] = f2bf(xv.x);
      Xt[(n4 + 1) * 34 + k] = f2bf(xv.y);
      Xt[(n4 + 2) * 34 + k] = f2bf(xv.z);
      Xt[(n4 + 3) * 34 + k] = f2bf(xv.w);
    }
    __syncthreads();
    v16bf a0 = load_frag_a(Wt + m0 * 32, 32);
    v16bf a1 = load_frag_a(Wt + (m0 + 16) * 32, 32);
#pragma unroll
    for (int j = 0; j < 4; ++j) {
      v16bf bb = load_frag_b(Xt + (n0w + 16 * j) * 34, 34);
      acc[0][j].v = wmma_bf16(a0, bb, acc[0][j].v);
      acc[1][j].v = wmma_bf16(a1, bb, acc[1][j].v);
    }
    __syncthreads();
  }
  const int lane = threadIdx.x & 31, ncol = lane & 15, half = lane >> 4;
#pragma unroll
  for (int i = 0; i < 2; ++i)
#pragma unroll
    for (int v = 0; v < 8; ++v) {
      int m = m0 + 16 * i + v + 8 * half;
      float bv = bias[m];
#pragma unroll
      for (int j = 0; j < 4; ++j) {
        int n = n0 + n0w + 16 * j + ncol;
        uint16_t h = f2bf(acc[i][j].f[v] + bv);
        if (p == 0) dst[(size_t)m * HW_ + n] = h;          // [CI][N]
        else        dst[(size_t)n * CI_ + m] = h;          // [N][CI]
      }
    }
}

// ---------------------------------------------------------------------------
// Kernel 2: fused dual-softmax flash attention.
// y[q,c] = 0.5*sum over both maps of softmax_k(s*Q K)[q,:] . g[c,:]
// Block = 64-query tile, 4 waves x 16 query rows; loops over 64-key tiles.
// All tile staging is contiguous b128 (async->LDS when available).
// ---------------------------------------------------------------------------
__global__ __launch_bounds__(128) void attn_kernel(
    const uint16_t* __restrict__ ws, uint16_t* __restrict__ ybuf_all) {
  const int qb = blockIdx.x, b = blockIdx.y;
  const size_t S = (size_t)BATCH * CI_ * HW_;
  const uint16_t* gb  = ws + 0 * S + (size_t)b * CI_ * HW_;  // [CI][N]
  const uint16_t* thT = ws + 1 * S + (size_t)b * CI_ * HW_;  // [N][CI]
  const uint16_t* phT = ws + 2 * S + (size_t)b * CI_ * HW_;
  const uint16_t* hqT = ws + 3 * S + (size_t)b * CI_ * HW_;
  const uint16_t* hkT = ws + 4 * S + (size_t)b * CI_ * HW_;
  uint16_t* ybufT = ybuf_all + (size_t)b * CI_ * HW_;        // [N][CI]
  const int q0 = qb * 64;
  const float scale = 0.08838834764831845f;  // 128^-0.5

  const int LDQ = 136, LDV = 72, LDP = 68;   // rows 16B-aligned for b128
  __shared__ __attribute__((aligned(16))) uint16_t K1[64 * 136];
  __shared__ __attribute__((aligned(16))) uint16_t K2[64 * 136];
  __shared__ __attribute__((aligned(16))) uint16_t Vg[128 * 72];
  __shared__ __attribute__((aligned(16))) uint16_t Pb[64 * 68];

  const int wave = threadIdx.x >> 5, wq0 = wave * 16;
  const int lane = threadIdx.x & 31, ncol = lane & 15, half = lane >> 4;

  // Stage Q tiles ([q][c], contiguous rows) via K1, hoist frags to regs.
  v16bf aq1[4], aq2[4];
  for (int i = threadIdx.x; i < 1024; i += 128) {
    int q = i >> 4, c8 = (i & 15) * 8;
    copy16_g2l(thT + (size_t)(q0 + q) * CI_ + c8, K1 + q * LDQ + c8);
  }
  copy_join();
  __syncthreads();
#pragma unroll
  for (int kc = 0; kc < 4; ++kc) aq1[kc] = load_frag_a(K1 + wq0 * LDQ + 32 * kc, LDQ);
  __syncthreads();
  for (int i = threadIdx.x; i < 1024; i += 128) {
    int q = i >> 4, c8 = (i & 15) * 8;
    copy16_g2l(hqT + (size_t)(q0 + q) * CI_ + c8, K1 + q * LDQ + c8);
  }
  copy_join();
  __syncthreads();
#pragma unroll
  for (int kc = 0; kc < 4; ++kc) aq2[kc] = load_frag_a(K1 + wq0 * LDQ + 32 * kc, LDQ);
  __syncthreads();

  AccT O1[8], O2[8];
  float m1[8], l1[8], m2[8], l2[8];
#pragma unroll
  for (int t = 0; t < 8; ++t)
#pragma unroll
    for (int e = 0; e < 8; ++e) { O1[t].f[e] = 0.f; O2[t].f[e] = 0.f; }
#pragma unroll
  for (int e = 0; e < 8; ++e) { m1[e] = -1e30f; l1[e] = 0.f; m2[e] = -1e30f; l2[e] = 0.f; }

  for (int kt = 0; kt < HW_ / 64; ++kt) {
    const int k0 = kt * 64;
    for (int i = threadIdx.x; i < 1024; i += 128) {
      int r = i >> 4, c8 = (i & 15) * 8;      // K tiles: [kl][c]
      copy16_g2l(phT + (size_t)(k0 + r) * CI_ + c8, K1 + r * LDQ + c8);
      copy16_g2l(hkT + (size_t)(k0 + r) * CI_ + c8, K2 + r * LDQ + c8);
    }
    for (int i = threadIdx.x; i < 1024; i += 128) {
      int c = i >> 3, k8 = (i & 7) * 8;       // V: [c][kl]
      copy16_g2l(gb + (size_t)c * HW_ + k0 + k8, Vg + c * LDV + k8);
    }
    copy_join();
    __syncthreads();

#pragma unroll
    for (int att = 0; att < 2; ++att) {
      const uint16_t* Kt = (att == 0) ? K1 : K2;
      const v16bf* aq = (att == 0) ? aq1 : aq2;
      float* mr = (att == 0) ? m1 : m2;
      float* lr = (att == 0) ? l1 : l2;
      AccT* O = (att == 0) ? O1 : O2;

      AccT s[4];
#pragma unroll
      for (int j = 0; j < 4; ++j) {
#pragma unroll
        for (int e = 0; e < 8; ++e) s[j].f[e] = 0.f;
#pragma unroll
        for (int kc = 0; kc < 4; ++kc) {
          v16bf bb = load_frag_b(Kt + (16 * j) * LDQ + 32 * kc, LDQ);
          s[j].v = wmma_bf16(aq[kc], bb, s[j].v);
        }
      }
      float corr[8];
#pragma unroll
      for (int v = 0; v < 8; ++v) {
        float mx = -1e30f;
#pragma unroll
        for (int j = 0; j < 4; ++j) {
          s[j].f[v] *= scale;
          mx = fmaxf(mx, s[j].f[v]);
        }
#pragma unroll
        for (int off = 1; off < 16; off <<= 1)
          mx = fmaxf(mx, __shfl_xor(mx, off, 32));
        float mnew = fmaxf(mr[v], mx);
        corr[v] = __expf(mr[v] - mnew);
        float rs = 0.f;
#pragma unroll
        for (int j = 0; j < 4; ++j) {
          float pv = __expf(s[j].f[v] - mnew);
          s[j].f[v] = pv;
          rs += pv;
        }
#pragma unroll
        for (int off = 1; off < 16; off <<= 1)
          rs += __shfl_xor(rs, off, 32);
        lr[v] = lr[v] * corr[v] + rs;
        mr[v] = mnew;
      }
      // P -> LDS (bf16) so it can be re-fragmented as the A operand of P@V.
#pragma unroll
      for (int v = 0; v < 8; ++v) {
        int q = wq0 + v + 8 * half;
#pragma unroll
        for (int j = 0; j < 4; ++j)
          Pb[q * LDP + 16 * j + ncol] = f2bf(s[j].f[v]);
      }
#pragma unroll
      for (int t = 0; t < 8; ++t)
#pragma unroll
        for (int v = 0; v < 8; ++v) O[t].f[v] *= corr[v];

      v16bf ap0 = load_frag_a(Pb + wq0 * LDP, LDP);
      v16bf ap1 = load_frag_a(Pb + wq0 * LDP + 32, LDP);
#pragma unroll
      for (int t = 0; t < 8; ++t) {
        v16bf b0 = load_frag_b(Vg + (16 * t) * LDV, LDV);
        v16bf b1 = load_frag_b(Vg + (16 * t) * LDV + 32, LDV);
        O[t].v = wmma_bf16(ap0, b0, O[t].v);
        O[t].v = wmma_bf16(ap1, b1, O[t].v);
      }
    }
    __syncthreads();
  }

  // Normalize, average, write y transposed [N][CI] (coalesced).
#pragma unroll
  for (int t = 0; t < 8; ++t)
#pragma unroll
    for (int v = 0; v < 8; ++v) {
      float val = 0.5f * (O1[t].f[v] / l1[v] + O2[t].f[v] / l2[v]);
      int c = 16 * t + ncol;
      int q = q0 + wq0 + v + 8 * half;
      ybufT[(size_t)q * CI_ + c] = f2bf(val);
    }
}

// ---------------------------------------------------------------------------
// Kernel 3: out = BN(W_w @ y + W_b); y stored [N][CI].
// ---------------------------------------------------------------------------
__global__ __launch_bounds__(256) void wout_kernel(
    const uint16_t* __restrict__ ybuf_all, const float* __restrict__ W_w,
    const float* __restrict__ W_b, const float* __restrict__ gamma,
    const float* __restrict__ beta, const float* __restrict__ mean,
    const float* __restrict__ var, float* __restrict__ out) {
  const int nb = blockIdx.x, b = blockIdx.y;
  const uint16_t* ybufT = ybuf_all + (size_t)b * CI_ * HW_;
  const int n0 = nb * 128;

  const int LDY = 40;                        // 16B-aligned rows
  __shared__ __attribute__((aligned(16))) uint16_t Wt[256 * 32];
  __shared__ __attribute__((aligned(16))) uint16_t Yt[128 * 40];

  const int wave = threadIdx.x >> 5;
  const int m0 = wave * 32;

  AccT acc[2][8];
#pragma unroll
  for (int i = 0; i < 2; ++i)
#pragma unroll
    for (int j = 0; j < 8; ++j)
#pragma unroll
      for (int e = 0; e < 8; ++e) acc[i][j].f[e] = 0.f;

  for (int kb = 0; kb < 4; ++kb) {
    for (int i = threadIdx.x; i < 2048; i += 256) {
      int m = i >> 3, k4 = (i & 7) * 4;
      float4 wv = *(const float4*)(W_w + m * CI_ + kb * 32 + k4);
      uint2 pk; pk.x = f2bf2(wv.x, wv.y); pk.y = f2bf2(wv.z, wv.w);
      *(uint2*)(Wt + m * 32 + k4) = pk;
    }
    for (int i = threadIdx.x; i < 512; i += 256) {
      int n = i >> 2, k8 = (i & 3) * 8;      // y rows contiguous in ci
      *(uint4*)(Yt + n * LDY + k8) =
          *(const uint4*)(ybufT + (size_t)(n0 + n) * CI_ + kb * 32 + k8);
    }
    __syncthreads();
    v16bf a0 = load_frag_a(Wt + m0 * 32, 32);
    v16bf a1 = load_frag_a(Wt + (m0 + 16) * 32, 32);
#pragma unroll
    for (int j = 0; j < 8; ++j) {
      v16bf bb = load_frag_b(Yt + (16 * j) * LDY, LDY);
      acc[0][j].v = wmma_bf16(a0, bb, acc[0][j].v);
      acc[1][j].v = wmma_bf16(a1, bb, acc[1][j].v);
    }
    __syncthreads();
  }
  const int lane = threadIdx.x & 31, ncol = lane & 15, half = lane >> 4;
#pragma unroll
  for (int i = 0; i < 2; ++i)
#pragma unroll
    for (int v = 0; v < 8; ++v) {
      int m = m0 + 16 * i + v + 8 * half;
      float inv = gamma[m] * rsqrtf(var[m] + EPS_);
      float sh = beta[m] - mean[m] * inv;
      float bi = W_b[m];
#pragma unroll
      for (int j = 0; j < 8; ++j)
        out[((size_t)b * C_IN + m) * HW_ + n0 + 16 * j + ncol] =
            (acc[i][j].f[v] + bi) * inv + sh;
    }
}

extern "C" void kernel_launch(void* const* d_in, const int* in_sizes, int n_in,
                              void* d_out, int out_size, void* d_ws, size_t ws_size,
                              hipStream_t stream) {
  (void)in_sizes; (void)n_in; (void)out_size; (void)ws_size;
  const float* feats = (const float*)d_in[0];
  const float* human = (const float*)d_in[1];
  const float* g_w  = (const float*)d_in[2];
  const float* g_b  = (const float*)d_in[3];
  const float* t_w  = (const float*)d_in[4];
  const float* t_b  = (const float*)d_in[5];
  const float* p_w  = (const float*)d_in[6];
  const float* p_b  = (const float*)d_in[7];
  const float* ht_w = (const float*)d_in[8];
  const float* ht_b = (const float*)d_in[9];
  const float* hp_w = (const float*)d_in[10];
  const float* hp_b = (const float*)d_in[11];
  const float* W_w  = (const float*)d_in[12];
  const float* W_b  = (const float*)d_in[13];
  const float* gamma = (const float*)d_in[14];
  const float* beta  = (const float*)d_in[15];
  const float* mean  = (const float*)d_in[16];
  const float* var   = (const float*)d_in[17];

  uint16_t* ws = (uint16_t*)d_ws;
  const size_t S = (size_t)BATCH * CI_ * HW_;
  uint16_t* ybuf = ws + 5 * S;  // workspace: 5 proj buffers + y, all bf16

  proj_kernel<<<dim3(HW_ / 128, BATCH, 5), 256, 0, stream>>>(
      feats, human, g_w, g_b, t_w, t_b, p_w, p_b, ht_w, ht_b, hp_w, hp_b, ws);
  attn_kernel<<<dim3(HW_ / 64, BATCH), 128, 0, stream>>>(ws, ybuf);
  wout_kernel<<<dim3(HW_ / 128, BATCH), 256, 0, stream>>>(
      ybuf, W_w, W_b, gamma, beta, mean, var, (float*)d_out);
}
